// LSTMEncoder_29996051595470
// MI455X (gfx1250) — matrix operand order repaired and checked by reading, not compiled, not run on hardware
//
#include <hip/hip_runtime.h>
#include <hip/hip_bf16.h>
#include <math.h>

// ---------------------------------------------------------------------------
// Problem constants
// ---------------------------------------------------------------------------
#define BATCH     32
#define SEQN      20
#define WDIM      512
#define HDIM      512
#define GDIM      2048       // 4*HDIM
#define ODIM      7680       // NT*SEM
#define NT        30
#define SEM       256
#define NPOS      640        // BATCH*SEQN
#define NROWS     6080       // BATCH * 190 window rows
#define CAT       1024       // 2*HDIM

typedef __attribute__((ext_vector_type(16))) __bf16 v16bf;
typedef __attribute__((ext_vector_type(8)))  __bf16 v8bf;
typedef __attribute__((ext_vector_type(8)))  float  v8f;

// off(k) = number of window-start rows before chunk k = (k-1)*(40-k)/2
__host__ __device__ inline int chunk_off(int k) { return ((k - 1) * (40 - k)) >> 1; }

__device__ inline float sigmf(float x) { return 1.0f / (1.0f + __expf(-x)); }

// ---------------------------------------------------------------------------
// WMMA tile loaders (wave32, 16x16x32 bf16)
// A tile 16x32 (MxK): lane<16 -> row M=lane, K 0..15;
//                     lane>=16 -> row M=lane-16, K 16..31
// ---------------------------------------------------------------------------
__device__ inline v16bf load_a_tile(const __bf16* base /* at [row0, k0] */,
                                    int lda, int lane) {
  int m   = lane & 15;
  int khi = lane >> 4;
  const __bf16* p = base + m * lda + khi * 16;
  v8bf lo = *(const v8bf*)(p);
  v8bf hi = *(const v8bf*)(p + 8);
  v16bf a;
#pragma unroll
  for (int i = 0; i < 8; ++i) { a[i] = lo[i]; a[i + 8] = hi[i]; }
  return a;
}

// B tile 32x16 (KxN) where mathematically B = W^T and W is row-major
// [out_col][k].  lane<16 -> N=lane, K 0..15 ; lane>=16 -> N=lane-16, K 16..31.
// That is 16 contiguous bf16 from one W row -> single 32B load.
__device__ inline v16bf load_bT_tile(const __bf16* Wbase /* at [col0, k0] */,
                                     int ldw, int lane) {
  int n  = lane & 15;
  int kb = (lane >> 4) * 16;
  return *(const v16bf*)(Wbase + n * ldw + kb);
}

// ---------------------------------------------------------------------------
// Kernel 1: embedding gather + length mask, f32 -> bf16
// ---------------------------------------------------------------------------
__global__ void k_embed(const int* __restrict__ caps,
                        const int* __restrict__ lens,
                        const float* __restrict__ emb,
                        __bf16* __restrict__ wv) {
  int bn = blockIdx.x;            // 0..639
  int b  = bn / SEQN;
  int n  = bn % SEQN;
  int tok = caps[bn];
  bool on = (n < lens[b]);
  for (int e = threadIdx.x; e < WDIM; e += blockDim.x) {
    float v = on ? emb[(size_t)tok * WDIM + e] : 0.0f;
    wv[(size_t)bn * WDIM + e] = (__bf16)v;
  }
}

// ---------------------------------------------------------------------------
// Kernel 2: f32 -> bf16 conversion (weights)
// ---------------------------------------------------------------------------
__global__ void k_cvt(const float* __restrict__ s, __bf16* __restrict__ d, int n) {
  int i = blockIdx.x * blockDim.x + threadIdx.x;
  if (i < n) d[i] = (__bf16)s[i];
}

// Kernel 3: bias = b_ih + b_hh
__global__ void k_bias(const float* __restrict__ a, const float* __restrict__ b,
                       float* __restrict__ o, int n) {
  int i = blockIdx.x * blockDim.x + threadIdx.x;
  if (i < n) o[i] = a[i] + b[i];
}

// ---------------------------------------------------------------------------
// Kernel 4: input projection  xproj[d] = wv @ W_ih_d^T + bias_d  (640x2048)
// grid (40, 16, 2), block 256 (8 waves); wave computes a 16x16 tile.
// ---------------------------------------------------------------------------
__global__ void k_xproj(const __bf16* __restrict__ wv,
                        const __bf16* __restrict__ wih0,
                        const __bf16* __restrict__ wih1,
                        const float*  __restrict__ bias0,
                        const float*  __restrict__ bias1,
                        float* __restrict__ xp0,
                        float* __restrict__ xp1) {
  int wave = threadIdx.x >> 5, lane = threadIdx.x & 31;
  int row0 = blockIdx.x * 16;
  int col0 = blockIdx.y * 128 + wave * 16;
  int dir  = blockIdx.z;
  const __bf16* W    = dir ? wih1 : wih0;
  const float*  bias = dir ? bias1 : bias0;
  float*        xp   = dir ? xp1 : xp0;

  v8f acc;
#pragma unroll
  for (int r = 0; r < 8; ++r) acc[r] = 0.0f;

  for (int k0 = 0; k0 < WDIM; k0 += 32) {
    v16bf a = load_a_tile(wv + (size_t)row0 * WDIM + k0, WDIM, lane);
    v16bf b = load_bT_tile(W + (size_t)col0 * WDIM + k0, WDIM, lane);
    acc = __builtin_amdgcn_wmma_f32_16x16x32_bf16(false, a, false, b,
                                                  (short)0, acc, false, false);
  }
  int n = lane & 15, mh = lane >> 4;
  float bv = bias[col0 + n];
#pragma unroll
  for (int r = 0; r < 8; ++r)
    xp[(size_t)(row0 + r + 8 * mh) * GDIM + col0 + n] = acc[r] + bv;
}

// ---------------------------------------------------------------------------
// Kernel 5: LSTM recurrence for window length k+1.
// grid (2*(20-k), 2) : x = group of 16 sequences, y = direction.
// Block = 256 threads = 8 waves; wave w owns hidden units [w*64, w*64+64).
// h lives in LDS (16x512 bf16); c-state & window-sum of h live in VGPRs.
// Per timestep: gates = xproj_row + h @ W_hh^T via 256 WMMAs per wave.
// ---------------------------------------------------------------------------
__global__ void __launch_bounds__(256)
k_lstm(const float*  __restrict__ xp0,
       const float*  __restrict__ xp1,
       const __bf16* __restrict__ whh0,
       const __bf16* __restrict__ whh1,
       __bf16* __restrict__ hs,
       int k, int base2) {
  int dir = blockIdx.y;
  const float*  xp  = dir ? xp1 : xp0;
  const __bf16* Whh = dir ? whh1 : whh0;
  int nk = SEQN - k;

  int wave = threadIdx.x >> 5, lane = threadIdx.x & 31;
  int n = lane & 15, mh = lane >> 4, kb = mh * 16;
  int s0 = blockIdx.x * 16;

  __shared__ __bf16 h_lds[16 * HDIM];
  for (int i = threadIdx.x; i < 16 * HDIM; i += blockDim.x)
    h_lds[i] = (__bf16)0.0f;
  __syncthreads();

  // xproj row base per output-row r (sequence s -> (b,w))
  int xrowbase[8];
#pragma unroll
  for (int r = 0; r < 8; ++r) {
    int s = s0 + r + 8 * mh;
    int b = s / nk, w = s % nk;
    xrowbase[r] = b * SEQN + w;
  }

  v8f cst[4], hsum[4];
#pragma unroll
  for (int j = 0; j < 4; ++j)
#pragma unroll
    for (int r = 0; r < 8; ++r) { cst[j][r] = 0.0f; hsum[j][r] = 0.0f; }

  for (int t = 0; t <= k; ++t) {
    int tpos = dir ? (k - t) : t;

    // init gate accumulators from precomputed input projection (+bias)
    v8f acc[16];
#pragma unroll
    for (int gi = 0; gi < 4; ++gi)
#pragma unroll
      for (int j = 0; j < 4; ++j) {
        int col = gi * HDIM + wave * 64 + j * 16 + n;
#pragma unroll
        for (int r = 0; r < 8; ++r)
          acc[gi * 4 + j][r] = xp[(size_t)(xrowbase[r] + tpos) * GDIM + col];
      }

    // gates += h @ W_hh^T   (16x512 * 512x2048 slice)
    for (int k0 = 0; k0 < HDIM; k0 += 32) {
      v16bf a = load_a_tile(&h_lds[k0], HDIM, lane);
#pragma unroll
      for (int tl = 0; tl < 16; ++tl) {
        int gi = tl >> 2, j = tl & 3;
        int col0 = gi * HDIM + wave * 64 + j * 16;
        const __bf16* bp = Whh + (size_t)(col0 + n) * HDIM + k0 + kb;
        __builtin_prefetch(bp + 32, 0, 1);   // global_prefetch_b8 for next K-step
        v16bf b = *(const v16bf*)bp;
        acc[tl] = __builtin_amdgcn_wmma_f32_16x16x32_bf16(
            false, a, false, b, (short)0, acc[tl], false, false);
      }
    }
    __syncthreads();   // all waves done reading h_lds

    // elementwise gate math; update c, h, window-sum; publish h
#pragma unroll
    for (int j = 0; j < 4; ++j) {
#pragma unroll
      for (int r = 0; r < 8; ++r) {
        float ig = acc[0 * 4 + j][r];
        float fg = acc[1 * 4 + j][r];
        float gg = acc[2 * 4 + j][r];
        float og = acc[3 * 4 + j][r];
        float c = sigmf(fg) * cst[j][r] + sigmf(ig) * tanhf(gg);
        float h = sigmf(og) * tanhf(c);
        cst[j][r]  = c;
        hsum[j][r] += h;
        int m = r + 8 * mh;
        int u = wave * 64 + j * 16 + n;
        h_lds[m * HDIM + u] = (__bf16)h;
      }
    }
    __syncthreads();
  }

  // store window-summed h (bf16) : forward -> cols [0,512), backward -> [512,1024)
#pragma unroll
  for (int j = 0; j < 4; ++j)
#pragma unroll
    for (int r = 0; r < 8; ++r) {
      int s   = s0 + r + 8 * mh;
      int col = dir * HDIM + wave * 64 + j * 16 + n;
      hs[(size_t)(base2 + s) * CAT + col] = (__bf16)hsum[j][r];
    }
}

// ---------------------------------------------------------------------------
// Kernel 6: output projection  out = hs @ w_out^T + (k+1)*b_out
// M=6080, N=7680, K=1024; grid (380, 60), block 256 (8 waves -> 128 cols).
// Writes straight into d_out with the chunk-concat row remapping.
// ---------------------------------------------------------------------------
__global__ void __launch_bounds__(256)
k_outproj(const __bf16* __restrict__ hs,
          const __bf16* __restrict__ wout,
          const float*  __restrict__ bout,
          float* __restrict__ out) {
  int wave = threadIdx.x >> 5, lane = threadIdx.x & 31;
  int row0 = blockIdx.x * 16;
  int col0 = blockIdx.y * 128 + wave * 16;
  int n = lane & 15, mh = lane >> 4, kb = mh * 16;

  v8f acc;
#pragma unroll
  for (int r = 0; r < 8; ++r) acc[r] = 0.0f;

  for (int k0 = 0; k0 < CAT; k0 += 32) {
    v16bf a = load_a_tile(hs + (size_t)row0 * CAT + k0, CAT, lane);
    const __bf16* bp = wout + (size_t)(col0 + n) * CAT + k0 + kb;
    __builtin_prefetch(bp + 32, 0, 1);
    v16bf b = *(const v16bf*)bp;
    acc = __builtin_amdgcn_wmma_f32_16x16x32_bf16(false, a, false, b,
                                                  (short)0, acc, false, false);
  }

  float bv = bout[col0 + n];
#pragma unroll
  for (int r = 0; r < 8; ++r) {
    int row = row0 + r + 8 * mh;        // [base2(k) + b*(20-k) + w]
    int kk = 1;
    while (kk < 19 && row >= BATCH * chunk_off(kk + 1)) ++kk;
    int offk = chunk_off(kk);
    int rem  = row - BATCH * offk;
    int nk   = SEQN - kk;
    int b    = rem / nk, w = rem % nk;
    int orow = b * 190 + offk + w;      // concat(chunks, axis=1) ordering
    out[(size_t)orow * ODIM + col0 + n] = acc[r] + (float)(kk + 1) * bv;
  }
}

// ---------------------------------------------------------------------------
// Kernel 7: in-place L2 normalize over SEM (256) per (row, nt)
// ---------------------------------------------------------------------------
__global__ void k_norm(float* __restrict__ out) {
  size_t base = (size_t)blockIdx.x * SEM;
  float v = out[base + threadIdx.x];
  float ss = v * v;
#pragma unroll
  for (int o = 16; o > 0; o >>= 1) ss += __shfl_xor(ss, o, 32);
  __shared__ float red[8];
  int wave = threadIdx.x >> 5, lane = threadIdx.x & 31;
  if (lane == 0) red[wave] = ss;
  __syncthreads();
  float tot = 0.0f;
#pragma unroll
  for (int i = 0; i < 8; ++i) tot += red[i];
  float nrm = fmaxf(sqrtf(tot), 1e-12f);
  out[base + threadIdx.x] = v / nrm;
}

// ---------------------------------------------------------------------------
// Host launch
// ---------------------------------------------------------------------------
extern "C" void kernel_launch(void* const* d_in, const int* in_sizes, int n_in,
                              void* d_out, int out_size, void* d_ws, size_t ws_size,
                              hipStream_t stream) {
  const int*   caps  = (const int*)d_in[1];
  const int*   lens  = (const int*)d_in[2];
  const float* emb   = (const float*)d_in[3];
  const float* wihf  = (const float*)d_in[4];
  const float* whhf  = (const float*)d_in[5];
  const float* bihf  = (const float*)d_in[6];
  const float* bhhf  = (const float*)d_in[7];
  const float* wihb  = (const float*)d_in[8];
  const float* whhb  = (const float*)d_in[9];
  const float* bihb  = (const float*)d_in[10];
  const float* bhhb  = (const float*)d_in[11];
  const float* woutf = (const float*)d_in[12];
  const float* boutf = (const float*)d_in[13];
  float* out = (float*)d_out;

  // workspace carving (~44 MB total)
  char* ws = (char*)d_ws;
  size_t off = 0;
  auto carve = [&](size_t bytes) -> void* {
    void* p = ws + off;
    off = (off + bytes + 255) & ~(size_t)255;
    return p;
  };
  __bf16* wv      = (__bf16*)carve((size_t)NPOS * WDIM * 2);
  __bf16* wih_b0  = (__bf16*)carve((size_t)GDIM * WDIM * 2);
  __bf16* wih_b1  = (__bf16*)carve((size_t)GDIM * WDIM * 2);
  __bf16* whh_b0  = (__bf16*)carve((size_t)GDIM * HDIM * 2);
  __bf16* whh_b1  = (__bf16*)carve((size_t)GDIM * HDIM * 2);
  __bf16* wout_b  = (__bf16*)carve((size_t)ODIM * CAT * 2);
  float*  bias0   = (float*)carve((size_t)GDIM * 4);
  float*  bias1   = (float*)carve((size_t)GDIM * 4);
  float*  xp0     = (float*)carve((size_t)NPOS * GDIM * 4);
  float*  xp1     = (float*)carve((size_t)NPOS * GDIM * 4);
  __bf16* hsbuf   = (__bf16*)carve((size_t)NROWS * CAT * 2);

  // 1. embeddings + mask
  k_embed<<<NPOS, 256, 0, stream>>>(caps, lens, emb, wv);

  // 2. weight conversions to bf16
  int nw = GDIM * WDIM;
  k_cvt<<<(nw + 255) / 256, 256, 0, stream>>>(wihf, wih_b0, nw);
  k_cvt<<<(nw + 255) / 256, 256, 0, stream>>>(wihb, wih_b1, nw);
  k_cvt<<<(nw + 255) / 256, 256, 0, stream>>>(whhf, whh_b0, nw);
  k_cvt<<<(nw + 255) / 256, 256, 0, stream>>>(whhb, whh_b1, nw);
  int no = ODIM * CAT;
  k_cvt<<<(no + 255) / 256, 256, 0, stream>>>(woutf, wout_b, no);

  // 3. combined biases
  k_bias<<<(GDIM + 255) / 256, 256, 0, stream>>>(bihf, bhhf, bias0, GDIM);
  k_bias<<<(GDIM + 255) / 256, 256, 0, stream>>>(bihb, bhhb, bias1, GDIM);

  // 4. input projection (both directions)
  k_xproj<<<dim3(NPOS / 16, GDIM / 128, 2), 256, 0, stream>>>(
      wv, wih_b0, wih_b1, bias0, bias1, xp0, xp1);

  // 5. LSTM recurrence per window length (both directions in grid.y)
  for (int k = 1; k < SEQN; ++k) {
    int nseq = BATCH * (SEQN - k);
    k_lstm<<<dim3(nseq / 16, 2), 256, 0, stream>>>(
        xp0, xp1, whh_b0, whh_b1, hsbuf, k, BATCH * chunk_off(k));
  }

  // 6. output projection + chunk-concat scatter + scaled bias
  k_outproj<<<dim3(NROWS / 16, ODIM / 128), 256, 0, stream>>>(
      hsbuf, wout_b, boutf, out);

  // 7. in-place normalize over SEM
  k_norm<<<NROWS * NT, 256, 0, stream>>>(out);
}
